// Model_68573447848305
// MI455X (gfx1250) — compile-verified, matrix operand order checked by reading, not run on hardware
//
#include <hip/hip_runtime.h>

// ---------------------------------------------------------------------------
// Types for CDNA5 WMMA
// ---------------------------------------------------------------------------
typedef __attribute__((ext_vector_type(16))) _Float16 v16h;
typedef __attribute__((ext_vector_type(8)))  _Float16 v8h;
typedef __attribute__((ext_vector_type(8)))  float    v8f;
typedef __attribute__((ext_vector_type(4)))  unsigned int v4u;

#define T_SEQ 128
#define B_TOT 1024
#define HID   128
#define G4    512           // 4*HID
#define HP    136           // padded LDS row stride (halfs) to dodge bank conflicts
#define WMAT_HALFS 65536    // 512*128 halfs per swizzled weight matrix

// Native gfx1250 v_tanh_f32 if the toolchain exposes it, else exp-based.
__device__ __forceinline__ float tanh_f(float x) {
#if __has_builtin(__builtin_amdgcn_tanhf)
    return __builtin_amdgcn_tanhf(x);
#elif __has_builtin(__builtin_amdgcn_tanh_f32)
    return __builtin_amdgcn_tanh_f32(x);
#else
    return 2.0f / (1.0f + __expf(-2.0f * x)) - 1.0f;
#endif
}
__device__ __forceinline__ float sigmoid_f(float x) {
    return 0.5f * tanh_f(0.5f * x) + 0.5f;   // 1 trans + 2 fma
}
__device__ __forceinline__ v16h mk16(v8h lo, v8h hi) {
    v16h r;
#pragma unroll
    for (int i = 0; i < 8; ++i) { r[i] = lo[i]; r[i + 8] = hi[i]; }
    return r;
}

// ---------------------------------------------------------------------------
// Weight swizzle: f32 [512,128] row-major  ->  f16 WMMA B-fragment order.
// dst[((kc*32 + ntile)*32 + lane)*16 + j]: each lane's fragment is 32
// contiguous bytes.  B[k][n] = W[n][k]; lane&15 = N%16;
// koff = (j&7) + 16*(j>>3) + 8*(lane>>4)   (ISA 7.12.2 16-bit layouts)
// ---------------------------------------------------------------------------
__global__ void swizzle_w_kernel(const float* __restrict__ src,
                                 _Float16* __restrict__ dst) {
    int idx = blockIdx.x * blockDim.x + threadIdx.x;   // 0..65535
    if (idx >= WMAT_HALFS) return;
    int j     = idx & 15;
    int lane  = (idx >> 4) & 31;
    int ntile = (idx >> 9) & 31;
    int kc    = idx >> 14;
    int n = ntile * 16 + (lane & 15);
    int k = kc * 32 + (j & 7) + 16 * (j >> 3) + 8 * (lane >> 4);
    dst[idx] = (_Float16)src[n * HID + k];
}

// ---------------------------------------------------------------------------
// One LSTM layer over the whole sequence.
// Grid = 64 blocks (16 batch rows each), 256 threads = 8 wave32.
// Wave w owns gate tiles {g*8+w : g=0..3} -> hidden cols [w*16, w*16+16).
// ALL weight B-fragments live in VGPRs (256 regs/wave; CDNA5 allows 1024).
// c-state lives in registers; h round-trips through a 4KB LDS tile.
// Inner loop: 16 ds_load_b128 (A-frags) -> 32 back-to-back WMMAs.
// ---------------------------------------------------------------------------
template <bool LAYER0>
__launch_bounds__(256, 1)
__global__ void lstm_layer_kernel(const _Float16* __restrict__ x_prev, // [T,B,H] f16 (null if LAYER0)
                                  const float* __restrict__ pv_in,     // prev_values [B,T] (LAYER0)
                                  const _Float16* __restrict__ wih_sw, // swizzled (null if LAYER0)
                                  const _Float16* __restrict__ whh_sw, // swizzled
                                  const float* __restrict__ wih0,      // [512] (LAYER0)
                                  const float* __restrict__ bias,      // [512]
                                  _Float16* __restrict__ ys_out)       // [T,B,H] f16
{
    __shared__ _Float16 lds_x[16 * HP];
    __shared__ _Float16 lds_h[16 * HP];

    const int tid   = threadIdx.x;
    const int wave  = tid >> 5;     // 0..7
    const int lane  = tid & 31;
    const int lhalf = lane >> 4;    // 0/1
    const int l15   = lane & 15;
    const int rowbase = blockIdx.x * 16;

    // ---- load this wave's weight B-fragments into registers (once) ----
    v16h whhf[4][4], wihf[4][4];
#pragma unroll
    for (int kc = 0; kc < 4; ++kc)
#pragma unroll
        for (int g = 0; g < 4; ++g) {
            const int fo = (((kc * 32 + (g * 8 + wave)) * 32) + lane) * 16;
            whhf[kc][g] = *(const v16h*)(whh_sw + fo);
            if (!LAYER0) wihf[kc][g] = *(const v16h*)(wih_sw + fo);
        }

    const int hcol = wave * 16 + l15;       // hidden column owned by this lane
    float bgate[4], w0g[4];
#pragma unroll
    for (int g = 0; g < 4; ++g) {
        bgate[g] = bias[g * HID + hcol];
        w0g[g]   = LAYER0 ? wih0[g * HID + hcol] : 0.0f;
    }

    v8f cst;                                  // c-state for (rows, hcol)
#pragma unroll
    for (int r = 0; r < 8; ++r) cst[r] = 0.0f;

    // zero the h tile
    for (int i = tid; i < 16 * HP; i += 256) lds_h[i] = (_Float16)0.0f;
    __syncthreads();

    for (int t = 0; t < T_SEQ; ++t) {
        // 1) stage x tile (contiguous 4KB for this batch tile) OR scalar xv
        float xv[8];
        if (LAYER0) {
#pragma unroll
            for (int r = 0; r < 8; ++r)
                xv[r] = pv_in[(size_t)(rowbase + r + 8 * lhalf) * T_SEQ + t];
        } else {
            const v4u* src = (const v4u*)(x_prev + (size_t)(t * B_TOT + rowbase) * HID);
            int row = tid >> 4, cg = (tid & 15) * 8;
            *(v4u*)(lds_x + row * HP + cg) = src[tid];
        }

        // 2) preload h A-fragments (h(t-1) valid since end-of-iter barrier)
        v16h ha[4];
#pragma unroll
        for (int kc = 0; kc < 4; ++kc) {
            const _Float16* p = lds_h + l15 * HP + kc * 32 + 8 * lhalf;
            ha[kc] = mk16(*(const v8h*)p, *(const v8h*)(p + 16));
        }

        __syncthreads();   // x tile visible; everyone done reading lds_h

        // 3) preload x A-fragments
        v16h xa[4];
        if (!LAYER0) {
#pragma unroll
            for (int kc = 0; kc < 4; ++kc) {
                const _Float16* p = lds_x + l15 * HP + kc * 32 + 8 * lhalf;
                xa[kc] = mk16(*(const v8h*)p, *(const v8h*)(p + 16));
            }
        }

        // 4) accumulators: bias (+ scalar x path for layer 0)
        v8f acc[4];
#pragma unroll
        for (int g = 0; g < 4; ++g)
#pragma unroll
            for (int r = 0; r < 8; ++r)
                acc[g][r] = LAYER0 ? (bgate[g] + xv[r] * w0g[g]) : bgate[g];

        // 5) GEMMs: all operands in registers -> 32 back-to-back WMMAs
        if (!LAYER0) {
#pragma unroll
            for (int kc = 0; kc < 4; ++kc)
#pragma unroll
                for (int g = 0; g < 4; ++g)
                    acc[g] = __builtin_amdgcn_wmma_f32_16x16x32_f16(
                        false, xa[kc], false, wihf[kc][g], (short)0, acc[g], false, false);
        }
#pragma unroll
        for (int kc = 0; kc < 4; ++kc)
#pragma unroll
            for (int g = 0; g < 4; ++g)
                acc[g] = __builtin_amdgcn_wmma_f32_16x16x32_f16(
                    false, ha[kc], false, whhf[kc][g], (short)0, acc[g], false, false);

        // 6) gates (i,f,g,o), update c-state, produce h
        v8f hnew;
#pragma unroll
        for (int r = 0; r < 8; ++r) {
            float iv = sigmoid_f(acc[0][r]);
            float fv = sigmoid_f(acc[1][r]);
            float gv = tanh_f(acc[2][r]);
            float ov = sigmoid_f(acc[3][r]);
            float cn = fv * cst[r] + iv * gv;
            cst[r] = cn;
            hnew[r] = ov * tanh_f(cn);
        }

        // 7) write h to LDS (next step's A) and to global ys
#pragma unroll
        for (int r = 0; r < 8; ++r) {
            int M = r + 8 * lhalf;
            _Float16 hv = (_Float16)hnew[r];
            lds_h[M * HP + hcol] = hv;
            ys_out[(size_t)(t * B_TOT + rowbase + M) * HID + hcol] = hv;
        }
        __syncthreads();   // h(t) complete
    }
}

// ---------------------------------------------------------------------------
// Head: pv = h@lin_w.T + lin_b, then feature concat [B,105]
// ---------------------------------------------------------------------------
__global__ void head_feat_kernel(const _Float16* __restrict__ hfin, // [B,128]
                                 const float* __restrict__ lin_w,   // [32,128]
                                 const float* __restrict__ lin_b,
                                 const int* __restrict__ day, const int* __restrict__ date,
                                 const int* __restrict__ grp, const int* __restrict__ gtin,
                                 const float* __restrict__ day_emb, const float* __restrict__ date_emb,
                                 const float* __restrict__ grp_emb, const float* __restrict__ gtin_emb,
                                 const float* __restrict__ price,
                                 float* __restrict__ feat) {
    int b = blockIdx.x, c = threadIdx.x;
    if (c >= 105) return;
    float v;
    if (c < 32) {
        float s = lin_b[c];
        const _Float16* h = hfin + (size_t)b * HID;
        for (int k = 0; k < HID; ++k) s += (float)h[k] * lin_w[c * HID + k];
        v = s;
    } else if (c < 36)  v = day_emb[day[b] * 4 + (c - 32)];
    else if (c < 40)    v = date_emb[date[b] * 4 + (c - 36)];
    else if (c < 72)    v = grp_emb[grp[b] * 32 + (c - 40)];
    else if (c < 104)   v = gtin_emb[gtin[b] * 32 + (c - 72)];
    else                v = price[b];
    feat[b * 105 + c] = v;
}

__global__ void head_dec1_kernel(const float* __restrict__ feat,
                                 const float* __restrict__ w1,  // [512,105]
                                 const float* __restrict__ b1,
                                 float* __restrict__ hid) {
    __shared__ float f[105];
    int b = blockIdx.x, j = threadIdx.x;   // 512 threads
    if (j < 105) f[j] = feat[b * 105 + j];
    __syncthreads();
    float s = b1[j];
    for (int k = 0; k < 105; ++k) s += f[k] * w1[j * 105 + k];
    hid[(size_t)b * 512 + j] = fmaxf(s, 0.0f);
}

__global__ void head_dec2_kernel(const float* __restrict__ hid,
                                 const float* __restrict__ w2,  // [512]
                                 const float* __restrict__ b2,
                                 float* __restrict__ out) {
    __shared__ float red[512];
    int b = blockIdx.x, j = threadIdx.x;   // 512 threads
    red[j] = hid[(size_t)b * 512 + j] * w2[j];
    __syncthreads();
    for (int s = 256; s > 0; s >>= 1) {
        if (j < s) red[j] += red[j + s];
        __syncthreads();
    }
    if (j == 0) out[b] = red[0] + b2[0];
}

// ---------------------------------------------------------------------------
// Launcher
// ---------------------------------------------------------------------------
extern "C" void kernel_launch(void* const* d_in, const int* in_sizes, int n_in,
                              void* d_out, int out_size, void* d_ws, size_t ws_size,
                              hipStream_t stream) {
    const float* prev_values = (const float*)d_in[0];
    const float* price       = (const float*)d_in[1];
    const int*   gtin        = (const int*)d_in[2];
    const int*   item_group  = (const int*)d_in[3];
    const int*   dow         = (const int*)d_in[4];
    const int*   date        = (const int*)d_in[5];
    const float* gtin_emb    = (const float*)d_in[6];
    const float* group_emb   = (const float*)d_in[7];
    const float* day_emb     = (const float*)d_in[8];
    const float* date_emb    = (const float*)d_in[9];
    const float* Wih0        = (const float*)d_in[10];
    const float* Whh0        = (const float*)d_in[11];
    const float* b0          = (const float*)d_in[12];
    const float* Wih         = (const float*)d_in[13];   // [11,512,128]
    const float* Whh         = (const float*)d_in[14];   // [11,512,128]
    const float* bvec        = (const float*)d_in[15];   // [11,512]
    const float* lin_w       = (const float*)d_in[16];
    const float* lin_b       = (const float*)d_in[17];
    const float* dec_w1      = (const float*)d_in[18];
    const float* dec_b1      = (const float*)d_in[19];
    const float* dec_w2      = (const float*)d_in[20];
    const float* dec_b2      = (const float*)d_in[21];

    // ---- workspace carve (~70 MB) ----
    size_t off = 0;
    auto carve = [&](size_t bytes) -> char* {
        char* p = (char*)d_ws + off;
        off += (bytes + 255) & ~(size_t)255;
        return p;
    };
    const size_t ys_bytes = (size_t)T_SEQ * B_TOT * HID * sizeof(_Float16); // 32 MB
    _Float16* ysA    = (_Float16*)carve(ys_bytes);
    _Float16* ysB    = (_Float16*)carve(ys_bytes);
    _Float16* whh_sw = (_Float16*)carve((size_t)12 * WMAT_HALFS * sizeof(_Float16));
    _Float16* wih_sw = (_Float16*)carve((size_t)11 * WMAT_HALFS * sizeof(_Float16));
    float*    feat   = (float*)carve((size_t)B_TOT * 105 * sizeof(float));
    float*    hid    = (float*)carve((size_t)B_TOT * 512 * sizeof(float));
    (void)ws_size; (void)in_sizes; (void)n_in; (void)out_size;

    // ---- weight swizzle (f32 -> f16 WMMA-fragment order) ----
    swizzle_w_kernel<<<256, 256, 0, stream>>>(Whh0, whh_sw);
    for (int l = 0; l < 11; ++l) {
        swizzle_w_kernel<<<256, 256, 0, stream>>>(Whh + (size_t)l * WMAT_HALFS,
                                                  whh_sw + (size_t)(l + 1) * WMAT_HALFS);
        swizzle_w_kernel<<<256, 256, 0, stream>>>(Wih + (size_t)l * WMAT_HALFS,
                                                  wih_sw + (size_t)l * WMAT_HALFS);
    }

    // ---- LSTM stack ----
    lstm_layer_kernel<true><<<B_TOT / 16, 256, 0, stream>>>(
        nullptr, prev_values, nullptr, whh_sw, Wih0, b0, ysA);

    const _Float16* xin = ysA;
    _Float16* xout = ysB;
    for (int l = 0; l < 11; ++l) {
        lstm_layer_kernel<false><<<B_TOT / 16, 256, 0, stream>>>(
            xin, nullptr,
            wih_sw + (size_t)l * WMAT_HALFS,
            whh_sw + (size_t)(l + 1) * WMAT_HALFS,
            nullptr, bvec + (size_t)l * G4, xout);
        const _Float16* tmp = xout; xout = (_Float16*)xin; xin = tmp;
    }

    // h_final = ys_last[T-1, :, :]
    const _Float16* hfin = xin + (size_t)(T_SEQ - 1) * B_TOT * HID;

    // ---- head ----
    head_feat_kernel<<<B_TOT, 128, 0, stream>>>(hfin, lin_w, lin_b,
                                                dow, date, item_group, gtin,
                                                day_emb, date_emb, group_emb, gtin_emb,
                                                price, feat);
    head_dec1_kernel<<<B_TOT, 512, 0, stream>>>(feat, dec_w1, dec_b1, hid);
    head_dec2_kernel<<<B_TOT, 512, 0, stream>>>(hid, dec_w2, dec_b2, (float*)d_out);
}
